// AttentionHead_74801150427402
// MI455X (gfx1250) — matrix operand order repaired and checked by reading, not compiled, not run on hardware
//
#include <hip/hip_runtime.h>

typedef __attribute__((ext_vector_type(16))) _Float16 v16h;
typedef __attribute__((ext_vector_type(8)))  _Float16 v8h;
typedef __attribute__((ext_vector_type(8)))  float    v8f;
typedef __attribute__((ext_vector_type(4)))  unsigned int u32x4;
typedef __attribute__((ext_vector_type(8)))  int          i32x8;
typedef __attribute__((ext_vector_type(4)))  int          i32x4;

#define BB  4
#define TT  4096
#define DIN 1024
#define DK  64
#define KVB 32          // keys per streamed block

static __device__ __forceinline__ v8f wmma_f16(v16h a, v16h b, v8f c) {
  // D = A(16x32 f16) * B(32x16 f16) + C(16x16 f32)
  return __builtin_amdgcn_wmma_f32_16x16x32_f16(
      /*neg_a=*/false, a, /*neg_b=*/false, b,
      /*c_mod=*/(short)0, c, /*reuse_a=*/false, /*reuse_b=*/false);
}

// Tensor Data Mover: 2D tile (f16 elements) global -> LDS.
// D# per CDNA5 ISA ch.8: group0 = {flags, lds_addr, global_addr, type=2},
// group1 = {data_size, tensor dims, tile dims, dim0 stride}. 2D only, so
// descriptor groups 2/3 (and the trailing group on this toolchain's 6-arg
// builtin) are zero.
static __device__ __forceinline__ void tdm_load_2d(
    const void* gaddr, unsigned lds_byte_off,
    unsigned tensor_d0, unsigned tensor_d1,
    unsigned tile_d0, unsigned tile_d1,
    unsigned long long stride0_elems)
{
  unsigned long long ga = (unsigned long long)(uintptr_t)gaddr;
  u32x4 g0;
  g0[0] = 0x1u;                                            // count=1, user mode
  g0[1] = lds_byte_off;                                    // lds_addr [63:32]
  g0[2] = (unsigned)(ga & 0xffffffffu);                    // global_addr lo
  g0[3] = (unsigned)((ga >> 32) & 0x01ffffffu) | (2u << 30); // ga hi | type=2
  i32x8 g1;
  g1[0] = (int)(1u << 16);                                 // data_size=1 -> 2B
  g1[1] = (int)((tensor_d0 & 0xffffu) << 16);              // tensor_dim0[15:0]
  g1[2] = (int)(((tensor_d0 >> 16) & 0xffffu) |
                ((tensor_d1 & 0xffffu) << 16));            // d0 hi | d1 lo
  g1[3] = (int)(((tensor_d1 >> 16) & 0xffffu) |
                ((tile_d0 & 0xffffu) << 16));              // d1 hi | tile_dim0
  g1[4] = (int)(tile_d1 & 0xffffu);                        // tile_dim1 (dim2=0)
  g1[5] = (int)(unsigned)(stride0_elems & 0xffffffffu);    // dim0_stride lo
  g1[6] = (int)(unsigned)((stride0_elems >> 32) & 0xffffu);// dim0_stride hi
  g1[7] = 0;
  i32x4 z4 = {0, 0, 0, 0};
  i32x8 z8 = {0, 0, 0, 0, 0, 0, 0, 0};
  __builtin_amdgcn_tensor_load_to_lds(g0, g1, z4, z4, z8, 0);
}

static __device__ __forceinline__ unsigned lds_off(const void* p) {
  return (unsigned)(unsigned long long)(uintptr_t)p;
}

// ---------------------------------------------------------------------------
// Projection: y = relu(x @ W^T + b), f16 out.
// One wave = one 16-token tile for one weight matrix (w = 0:K, 1:Q, 2:V).
// Q,K row-major [B*T][64]; V transposed [B][64][T].
// ---------------------------------------------------------------------------
__global__ __launch_bounds__(256) void proj_kernel(
    const float* __restrict__ x,
    const float* __restrict__ Wk, const float* __restrict__ bk,
    const float* __restrict__ Wq, const float* __restrict__ bq,
    const float* __restrict__ Wv, const float* __restrict__ bv,
    _Float16* __restrict__ Qh, _Float16* __restrict__ Kh,
    _Float16* __restrict__ Vt)
{
  const int gw   = blockIdx.x * 8 + (threadIdx.x >> 5);  // 0..3071
  const int lane = threadIdx.x & 31;
  const int w    = gw % 3;
  const int tile = gw / 3;            // 0..1023
  const int t0   = tile * 16;

  const float* W;  const float* bias;
  if      (w == 0) { W = Wk; bias = bk; }
  else if (w == 1) { W = Wq; bias = bq; }
  else             { W = Wv; bias = bv; }

  const int half = lane >> 4;
  const int col  = lane & 15;

  const v8f zero = {};
  v8f acc[4] = {zero, zero, zero, zero};

  // A 16x32 f16 layout: lane holds row (lane&15); k = 16*half .. +15.
  const float* xrow  = x + (size_t)(t0 + col) * DIN;
  const int    a_off = half * 16;

  for (int db = 0; db < DIN / 32; ++db) {
    const int dbase = db * 32;
    v16h a;
#pragma unroll
    for (int i = 0; i < 8; ++i) a[i]     = (_Float16)xrow[dbase + a_off + i];
#pragma unroll
    for (int i = 0; i < 8; ++i) a[8 + i] = (_Float16)xrow[dbase + a_off + 8 + i];

    // B 32x16: lane holds column h = hb*16+col, k = 16*half .. +15.
#pragma unroll
    for (int hb = 0; hb < 4; ++hb) {
      const float* wrow = W + (size_t)(hb * 16 + col) * DIN + dbase + half * 16;
      v16h bm;
#pragma unroll
      for (int i = 0; i < 16; ++i) bm[i] = (_Float16)wrow[i];
      acc[hb] = wmma_f16(a, bm, acc[hb]);
    }
  }

  // C/D layout: VGPR j -> row (j + 8*half), lane -> col.
#pragma unroll
  for (int hb = 0; hb < 4; ++hb) {
    const int   h   = hb * 16 + col;
    const float bv_ = bias[h];
#pragma unroll
    for (int j = 0; j < 8; ++j) {
      const int      trow = t0 + j + 8 * half;
      const float    vv   = fmaxf(acc[hb][j] + bv_, 0.0f);
      const _Float16 hv   = (_Float16)vv;
      if      (w == 0) Kh[(size_t)trow * DK + h] = hv;
      else if (w == 1) Qh[(size_t)trow * DK + h] = hv;
      else {
        const int bI = trow / TT, tl = trow % TT;
        Vt[((size_t)bI * DK + h) * TT + tl] = hv;
      }
    }
  }
}

// ---------------------------------------------------------------------------
// Flash attention. 8 waves/block = 8 query tiles (same batch) sharing each
// 32-key K/V block in LDS, double-buffered via the Tensor Data Mover.
// ---------------------------------------------------------------------------
__global__ __launch_bounds__(256) void attn_kernel(
    const _Float16* __restrict__ Qh, const _Float16* __restrict__ Kh,
    const _Float16* __restrict__ Vt, float* __restrict__ out)
{
  __shared__ _Float16 Kbuf[2][KVB * DK];   // [32 keys][64 feat]   2 x 4 KB
  __shared__ _Float16 Vbuf[2][DK * KVB];   // [64 vcol][32 keys]   2 x 4 KB
  __shared__ _Float16 Pbuf[8][16 * 32];    // per-wave P tile      8 KB

  const int wv   = threadIdx.x >> 5;
  const int lane = threadIdx.x & 31;
  const int tile = blockIdx.x * 8 + wv;    // 0..1023
  const int b    = blockIdx.x >> 5;        // 32 blocks per batch
  const int q0   = tile * 16;              // global query token
  const int half = lane >> 4;
  const int col  = lane & 15;

  // Q A-tiles resident for the whole pass.
  const _Float16* qrow = Qh + (size_t)(q0 + col) * DK;
  v16h qa0, qa1;
  {
    v8h c0 = *(const v8h*)(qrow + 16 * half);
    v8h c1 = *(const v8h*)(qrow + 16 * half + 8);
    v8h c2 = *(const v8h*)(qrow + 32 + 16 * half);
    v8h c3 = *(const v8h*)(qrow + 32 + 16 * half + 8);
#pragma unroll
    for (int i = 0; i < 8; ++i) {
      qa0[i] = c0[i]; qa0[8 + i] = c1[i];
      qa1[i] = c2[i]; qa1[8 + i] = c3[i];
    }
  }

  const v8f zero = {};
  v8f o[4] = {zero, zero, zero, zero};
  float m[8], l[8];
#pragma unroll
  for (int j = 0; j < 8; ++j) { m[j] = -1e30f; l[j] = 0.0f; }

  const _Float16* Kg = Kh + (size_t)b * TT * DK;   // [T][64]
  const _Float16* Vg = Vt + (size_t)b * DK * TT;   // [64][T]
  _Float16* pl = Pbuf[wv];

  const int NBLK = TT / KVB;   // 128

  // Preload block 0 (wave 0 drives the TDM; branch is wave-uniform).
  if (wv == 0) {
    tdm_load_2d(Kg, lds_off(Kbuf[0]), DK, TT, DK, KVB, DK);
    tdm_load_2d(Vg, lds_off(Vbuf[0]), TT, DK, KVB, DK, TT);
  }

  for (int i = 0; i < NBLK; ++i) {
    const int cur = i & 1;
    if (wv == 0) __builtin_amdgcn_s_wait_tensorcnt(0);  // buf[cur] landed
    __syncthreads();   // publish buf[cur]; everyone done reading buf[cur^1]

    if (wv == 0 && (i + 1) < NBLK) {                    // overlap next DMA
      const int kb2 = (i + 1) * KVB;
      tdm_load_2d(Kg + (size_t)kb2 * DK, lds_off(Kbuf[cur ^ 1]),
                  DK, TT, DK, KVB, DK);
      tdm_load_2d(Vg + kb2, lds_off(Vbuf[cur ^ 1]),
                  TT, DK, KVB, DK, TT);
    }

    const _Float16* KB = Kbuf[cur];
    const _Float16* VB = Vbuf[cur];

    // ---- scores: two 16x16 tiles, contraction over DK=64, B from LDS ----
    v8f s0 = zero, s1 = zero;
    {
      const _Float16* kr0 = KB + (size_t)col * DK + 16 * half;        // keys 0..15
      v16h bm0 = *(const v16h*)(kr0);
      v16h bm1 = *(const v16h*)(kr0 + 32);
      s0 = wmma_f16(qa0, bm0, s0);
      s0 = wmma_f16(qa1, bm1, s0);
      const _Float16* kr1 = KB + (size_t)(16 + col) * DK + 16 * half; // keys 16..31
      v16h bm2 = *(const v16h*)(kr1);
      v16h bm3 = *(const v16h*)(kr1 + 32);
      s1 = wmma_f16(qa0, bm2, s1);
      s1 = wmma_f16(qa1, bm3, s1);
    }

    // ---- online softmax (row = j + 8*half striped across half-wave) ----
    const float scale = 0.125f;  // 1/sqrt(64)
    float p0[8], p1[8];
#pragma unroll
    for (int j = 0; j < 8; ++j) {
      float a0 = s0[j] * scale, a1 = s1[j] * scale;
      float mx = fmaxf(a0, a1);
      mx = fmaxf(mx, __shfl_xor(mx, 1, 32));
      mx = fmaxf(mx, __shfl_xor(mx, 2, 32));
      mx = fmaxf(mx, __shfl_xor(mx, 4, 32));
      mx = fmaxf(mx, __shfl_xor(mx, 8, 32));
      const float mn = fmaxf(m[j], mx);
      const float al = __expf(m[j] - mn);
      m[j] = mn;
      p0[j] = __expf(a0 - mn);
      p1[j] = __expf(a1 - mn);
      float rs = p0[j] + p1[j];
      rs += __shfl_xor(rs, 1, 32);
      rs += __shfl_xor(rs, 2, 32);
      rs += __shfl_xor(rs, 4, 32);
      rs += __shfl_xor(rs, 8, 32);
      l[j] = l[j] * al + rs;
#pragma unroll
      for (int n = 0; n < 4; ++n) o[n][j] *= al;
    }

    // ---- C-layout -> A-layout transpose of P through per-wave LDS ----
#pragma unroll
    for (int j = 0; j < 8; ++j) {
      const int r = j + 8 * half;
      pl[r * 32 + col]      = (_Float16)p0[j];
      pl[r * 32 + 16 + col] = (_Float16)p1[j];
    }
    asm volatile("s_wait_dscnt 0" ::: "memory");

    v16h pA;
    {
      const _Float16* prow = pl + col * 32;
      v8h a0 = *(const v8h*)(prow + 16 * half);
      v8h a1 = *(const v8h*)(prow + 16 * half + 8);
#pragma unroll
      for (int i2 = 0; i2 < 8; ++i2) { pA[i2] = a0[i2]; pA[8 + i2] = a1[i2]; }
    }

    // ---- O += P(16x32) @ Vblock(32x64): B columns contiguous in VB ----
#pragma unroll
    for (int n = 0; n < 4; ++n) {
      const _Float16* vrow = VB + (size_t)(n * 16 + col) * KVB + 16 * half;
      v16h bmv = *(const v16h*)vrow;
      o[n] = wmma_f16(pA, bmv, o[n]);
    }
  }

  // ---- normalize and write out (f32) ----
#pragma unroll
  for (int j = 0; j < 8; ++j) {
    const float inv = 1.0f / l[j];
    const int   t   = q0 + j + 8 * half;
#pragma unroll
    for (int n = 0; n < 4; ++n)
      out[(size_t)t * DK + n * 16 + col] = o[n][j] * inv;
  }
}

// ---------------------------------------------------------------------------
extern "C" void kernel_launch(void* const* d_in, const int* in_sizes, int n_in,
                              void* d_out, int out_size, void* d_ws, size_t ws_size,
                              hipStream_t stream) {
  const float* x  = (const float*)d_in[0];
  const float* Wk = (const float*)d_in[1];
  const float* bk = (const float*)d_in[2];
  const float* Wq = (const float*)d_in[3];
  const float* bq = (const float*)d_in[4];
  const float* Wv = (const float*)d_in[5];
  const float* bv = (const float*)d_in[6];

  char* ws = (char*)d_ws;
  const size_t qk_bytes = (size_t)BB * TT * DK * sizeof(short);  // 2 MB each
  _Float16* Qh = (_Float16*)(ws);
  _Float16* Kh = (_Float16*)(ws + qk_bytes);
  _Float16* Vt = (_Float16*)(ws + 2 * qk_bytes);

  proj_kernel<<<384, 256, 0, stream>>>(x, Wk, bk, Wq, bq, Wv, bv, Qh, Kh, Vt);
  attn_kernel<<<128, 256, 0, stream>>>(Qh, Kh, Vt, (float*)d_out);
}
